// LSTMWithTransformer_30270929502756
// MI455X (gfx1250) — compile-verified, matrix operand order checked by reading, not run on hardware
//
#include <hip/hip_runtime.h>
#include <hip/hip_bf16.h>

// Problem dims (match reference)
#define B_ 256
#define T_ 128
#define I_ 512
#define E_ 1024
#define F_ 2048
#define O_ 22
#define L_ 6

// LDS row pitch in bf16 elements for a 64-wide k tile: 72 elems = 144B.
// Row offsets stay 16B-aligned (144 = 9*16) and rows spread across banks
// (stride 36 dwords, r*36 mod 64 distinct for r=0..15).
#define SMP 72

typedef __attribute__((ext_vector_type(16))) __bf16 bf16x16;
typedef __attribute__((ext_vector_type(8)))  __bf16 bf16x8;
typedef __attribute__((ext_vector_type(8)))  float  f32x8;

union Frag { bf16x16 v; bf16x8 h[2]; };

__device__ __forceinline__ bf16x8 cvt8(const float4& a, const float4& b) {
    bf16x8 v;
    v[0] = (__bf16)a.x; v[1] = (__bf16)a.y; v[2] = (__bf16)a.z; v[3] = (__bf16)a.w;
    v[4] = (__bf16)b.x; v[5] = (__bf16)b.y; v[6] = (__bf16)b.z; v[7] = (__bf16)b.w;
    return v;
}

__device__ __forceinline__ f32x8 wmma_bf16(const Frag& a, const Frag& b, f32x8 c) {
    return __builtin_amdgcn_wmma_f32_16x16x32_bf16(
        /*neg_a=*/false, a.v, /*neg_b=*/false, b.v,
        /*c_mod=*/(short)0, c, /*reuse_a=*/false, /*reuse_b=*/false);
}

__device__ __forceinline__ float sigmoidf_(float x) {
    return 1.0f / (1.0f + __expf(-x));
}

// ---------------------------------------------------------------------------
// Split-K GEMM partials: Cpart[z][m][n] = sum_{k in z-th K-slice} A[m,k]*W[n,k]
// A: (M x K) row-major (lda); W: (N x K) row-major (torch x@W.T layout) -- the
// WMMA B operand is B^T in A-layout, so both operands use the same fragment
// pattern. Block = 256 thr / 8 waves, tile 64x64, double-buffered 64-k LDS
// stages (one barrier per stage; next-stage global loads overlap WMMAs).
// Wave w: M-tile = w&3, N-tiles = (w>>2)*2 + {0,1}.
// ---------------------------------------------------------------------------
__global__ __launch_bounds__(256)
void gemm_kernel(const float* __restrict__ A, int lda,
                 const float* __restrict__ W,
                 float* __restrict__ Cpart,
                 int N, int K, int klen)
{
    __shared__ __align__(16) __bf16 smA[2][64 * SMP];
    __shared__ __align__(16) __bf16 smW[2][64 * SMP];

    const int tid  = threadIdx.x;
    const int lane = tid & 31;
    const int wid  = tid >> 5;
    const int m0   = blockIdx.x * 64;
    const int n0   = blockIdx.y * 64;
    const int kz0  = blockIdx.z * klen;
    const int mt   = wid & 3;
    const int ng   = wid >> 2;
    const int r16  = lane & 15;
    const int kr   = (lane >> 4) << 3;
    const int mhi  = (lane >> 4) << 3;

    // Cooperative-load mapping: each thread owns (row, row+32) x one 8-elem oct.
    const int lrow = tid >> 3;     // 0..31
    const int oct  = tid & 7;      // 0..7
    const int ar0 = m0 + lrow, ar1 = m0 + 32 + lrow;
    int wr0 = n0 + lrow;      if (wr0 > N - 1) wr0 = N - 1;   // N=22 head clamp
    int wr1 = n0 + 32 + lrow; if (wr1 > N - 1) wr1 = N - 1;

    f32x8 acc0 = {}; f32x8 acc1 = {};

    // Prologue: stage 0 -> buffer 0
    {
        const float* pA0 = A + (size_t)ar0 * lda + kz0 + oct * 8;
        const float* pA1 = A + (size_t)ar1 * lda + kz0 + oct * 8;
        const float* pW0 = W + (size_t)wr0 * K + kz0 + oct * 8;
        const float* pW1 = W + (size_t)wr1 * K + kz0 + oct * 8;
        *(bf16x8*)&smA[0][lrow * SMP + oct * 8]        = cvt8(((const float4*)pA0)[0], ((const float4*)pA0)[1]);
        *(bf16x8*)&smA[0][(32 + lrow) * SMP + oct * 8] = cvt8(((const float4*)pA1)[0], ((const float4*)pA1)[1]);
        *(bf16x8*)&smW[0][lrow * SMP + oct * 8]        = cvt8(((const float4*)pW0)[0], ((const float4*)pW0)[1]);
        *(bf16x8*)&smW[0][(32 + lrow) * SMP + oct * 8] = cvt8(((const float4*)pW1)[0], ((const float4*)pW1)[1]);
    }
    __syncthreads();

    const int nstage = klen >> 6;
    for (int s = 0; s < nstage; ++s) {
        const int cur = s & 1;
        const bool more = (s + 1) < nstage;
        float4 pa0[2], pa1[2], pw0[2], pw1[2];
        if (more) {   // issue next-stage global loads before this stage's WMMAs
            const int kb = kz0 + (s + 1) * 64 + oct * 8;
            const float* pA0 = A + (size_t)ar0 * lda + kb;
            const float* pA1 = A + (size_t)ar1 * lda + kb;
            const float* pW0 = W + (size_t)wr0 * K + kb;
            const float* pW1 = W + (size_t)wr1 * K + kb;
            pa0[0] = ((const float4*)pA0)[0]; pa0[1] = ((const float4*)pA0)[1];
            pa1[0] = ((const float4*)pA1)[0]; pa1[1] = ((const float4*)pA1)[1];
            pw0[0] = ((const float4*)pW0)[0]; pw0[1] = ((const float4*)pW0)[1];
            pw1[0] = ((const float4*)pW1)[0]; pw1[1] = ((const float4*)pW1)[1];
        }
        const __bf16* sa = smA[cur];
        const __bf16* sw = smW[cur];
#pragma unroll
        for (int ks = 0; ks < 64; ks += 32) {
            Frag a, b0, b1;
            a.h[0]  = *(const bf16x8*)&sa[(mt * 16 + r16) * SMP + ks + kr];
            a.h[1]  = *(const bf16x8*)&sa[(mt * 16 + r16) * SMP + ks + 16 + kr];
            b0.h[0] = *(const bf16x8*)&sw[((ng * 2 + 0) * 16 + r16) * SMP + ks + kr];
            b0.h[1] = *(const bf16x8*)&sw[((ng * 2 + 0) * 16 + r16) * SMP + ks + 16 + kr];
            b1.h[0] = *(const bf16x8*)&sw[((ng * 2 + 1) * 16 + r16) * SMP + ks + kr];
            b1.h[1] = *(const bf16x8*)&sw[((ng * 2 + 1) * 16 + r16) * SMP + ks + 16 + kr];
            acc0 = wmma_bf16(a, b0, acc0);
            acc1 = wmma_bf16(a, b1, acc1);
        }
        if (more) {
            const int nxt = cur ^ 1;
            *(bf16x8*)&smA[nxt][lrow * SMP + oct * 8]        = cvt8(pa0[0], pa0[1]);
            *(bf16x8*)&smA[nxt][(32 + lrow) * SMP + oct * 8] = cvt8(pa1[0], pa1[1]);
            *(bf16x8*)&smW[nxt][lrow * SMP + oct * 8]        = cvt8(pw0[0], pw0[1]);
            *(bf16x8*)&smW[nxt][(32 + lrow) * SMP + oct * 8] = cvt8(pw1[0], pw1[1]);
        }
        __syncthreads();
    }

    // Raw f32 partials (bias/act/residual applied in combine pass).
    float* out = Cpart + (size_t)blockIdx.z * B_ * N;
#pragma unroll
    for (int t = 0; t < 2; ++t) {
        f32x8 acc = t ? acc1 : acc0;
        int n = n0 + (ng * 2 + t) * 16 + r16;
        if (n < N) {
#pragma unroll
            for (int r = 0; r < 8; ++r) {
                int m = m0 + mt * 16 + mhi + r;
                out[(size_t)m * N + n] = acc[r];
            }
        }
    }
}

// Combine split-K partials; fuse bias + optional ReLU + optional residual.
__global__ __launch_bounds__(256)
void combine_kernel(const float* __restrict__ part, const float* __restrict__ bias,
                    const float* resid, float* C, int N, int S, int relu)
{
    int idx = blockIdx.x * 256 + threadIdx.x;
    if (idx < B_ * N) {
        int n = idx % N;
        float v = bias[n];
        for (int z = 0; z < S; ++z) v += part[(size_t)z * B_ * N + idx];
        if (relu) v = fmaxf(v, 0.0f);
        if (resid) v += resid[idx];
        C[idx] = v;
    }
}

// ---------------------------------------------------------------------------
// Fused LSTM step: gates = x_t@Wih^T + h@Whh^T + b; c,h update.
// Block tile: 32 batch rows x 32 E cols x 4 gates; 8 waves: wave w = gate w>>1,
// M-half w&1, 2 N-tiles. Double-buffered 64-k LDS stages over both K phases.
// Gate pre-activations exchanged through LDS for the cell update.
// ---------------------------------------------------------------------------
__global__ __launch_bounds__(256)
void lstm_step_kernel(const float* __restrict__ x, int t,
                      const float* __restrict__ h_in,
                      float* __restrict__ h_out,
                      float* __restrict__ cst,
                      const float* __restrict__ wih,
                      const float* __restrict__ whh,
                      const float* __restrict__ bih,
                      const float* __restrict__ bhh)
{
    __shared__ __align__(16) __bf16 smA[2][32 * SMP];        //  9.2 KB
    __shared__ __align__(16) __bf16 smW[2][4][32 * SMP];     // 36.9 KB
    __shared__ float smG[4][32][33];                         // 16.9 KB

    const int tid  = threadIdx.x;
    const int lane = tid & 31;
    const int wid  = tid >> 5;
    const int gate = wid >> 1;   // 0=i 1=f 2=g 3=o (torch order)
    const int mt   = wid & 1;
    const int m0   = blockIdx.x * 32;
    const int n0   = blockIdx.y * 32;
    const int r16  = lane & 15;
    const int kr   = (lane >> 4) << 3;
    const int mhi  = (lane >> 4) << 3;
    const int lrow = tid >> 3;   // 0..31
    const int oct  = tid & 7;    // 0..7

    f32x8 acc0 = {}; f32x8 acc1 = {};

    for (int ph = 0; ph < 2; ++ph) {
        const float* Ap  = ph ? h_in : (x + (size_t)t * I_);
        const size_t lda = ph ? (size_t)E_ : (size_t)T_ * I_;
        const float* Wp  = ph ? whh : wih;
        const int    K   = ph ? E_ : I_;
        const size_t aOff = (size_t)(m0 + lrow) * lda + oct * 8;

        __syncthreads();   // phase boundary: previous buffer reads done
        {   // prologue: stage 0 -> buffer 0
            const float* pA = Ap + aOff;
            *(bf16x8*)&smA[0][lrow * SMP + oct * 8] =
                cvt8(((const float4*)pA)[0], ((const float4*)pA)[1]);
#pragma unroll
            for (int g = 0; g < 4; ++g) {
                const float* p = Wp + (size_t)(g * E_ + n0 + lrow) * K + oct * 8;
                *(bf16x8*)&smW[0][g][lrow * SMP + oct * 8] =
                    cvt8(((const float4*)p)[0], ((const float4*)p)[1]);
            }
        }
        __syncthreads();

        const int nstage = K >> 6;
        for (int s = 0; s < nstage; ++s) {
            const int cur = s & 1;
            const bool more = (s + 1) < nstage;
            float4 pa[2], pw[4][2];
            if (more) {
                const int kb = (s + 1) * 64;
                const float* pA = Ap + aOff + kb;
                pa[0] = ((const float4*)pA)[0]; pa[1] = ((const float4*)pA)[1];
#pragma unroll
                for (int g = 0; g < 4; ++g) {
                    const float* p = Wp + (size_t)(g * E_ + n0 + lrow) * K + oct * 8 + kb;
                    pw[g][0] = ((const float4*)p)[0]; pw[g][1] = ((const float4*)p)[1];
                }
            }
            const __bf16* sa = smA[cur];
            const __bf16* sw = smW[cur][gate];
#pragma unroll
            for (int ks = 0; ks < 64; ks += 32) {
                Frag a, b0, b1;
                a.h[0]  = *(const bf16x8*)&sa[(mt * 16 + r16) * SMP + ks + kr];
                a.h[1]  = *(const bf16x8*)&sa[(mt * 16 + r16) * SMP + ks + 16 + kr];
                b0.h[0] = *(const bf16x8*)&sw[r16 * SMP + ks + kr];
                b0.h[1] = *(const bf16x8*)&sw[r16 * SMP + ks + 16 + kr];
                b1.h[0] = *(const bf16x8*)&sw[(16 + r16) * SMP + ks + kr];
                b1.h[1] = *(const bf16x8*)&sw[(16 + r16) * SMP + ks + 16 + kr];
                acc0 = wmma_bf16(a, b0, acc0);
                acc1 = wmma_bf16(a, b1, acc1);
            }
            if (more) {
                const int nxt = cur ^ 1;
                *(bf16x8*)&smA[nxt][lrow * SMP + oct * 8] = cvt8(pa[0], pa[1]);
#pragma unroll
                for (int g = 0; g < 4; ++g)
                    *(bf16x8*)&smW[nxt][g][lrow * SMP + oct * 8] = cvt8(pw[g][0], pw[g][1]);
            }
            __syncthreads();
        }
    }

    // Exchange gate pre-activations through LDS.
#pragma unroll
    for (int r = 0; r < 8; ++r) {
        smG[gate][mt * 16 + mhi + r][r16]      = acc0[r];
        smG[gate][mt * 16 + mhi + r][16 + r16] = acc1[r];
    }
    __syncthreads();

    // Cell update: 32x32 outputs, 4 per thread.
#pragma unroll
    for (int i = 0; i < 4; ++i) {
        int idx = tid + i * 256;
        int ml = idx >> 5, nl = idx & 31;
        int n = n0 + nl;
        size_t off = (size_t)(m0 + ml) * E_ + n;
        float gi = smG[0][ml][nl] + bih[n]          + bhh[n];
        float gf = smG[1][ml][nl] + bih[E_ + n]     + bhh[E_ + n];
        float gc = smG[2][ml][nl] + bih[2 * E_ + n] + bhh[2 * E_ + n];
        float go = smG[3][ml][nl] + bih[3 * E_ + n] + bhh[3 * E_ + n];
        float cv = cst[off];
        cv = sigmoidf_(gf) * cv + sigmoidf_(gi) * tanhf(gc);
        cst[off]   = cv;
        h_out[off] = sigmoidf_(go) * tanhf(cv);
    }
}

// ---------------------------------------------------------------------------
// LayerNorm over last dim N; one block per row; safe in place.
// ---------------------------------------------------------------------------
__global__ __launch_bounds__(256)
void ln_kernel(const float* X, const float* g, const float* bta, float* Y, int N)
{
    __shared__ float red[256];
    const int row = blockIdx.x, tid = threadIdx.x;
    const float* xr = X + (size_t)row * N;

    float s = 0.f;
    for (int i = tid; i < N; i += 256) s += xr[i];
    red[tid] = s; __syncthreads();
    for (int st = 128; st; st >>= 1) { if (tid < st) red[tid] += red[tid + st]; __syncthreads(); }
    const float mu = red[0] * (1.0f / N);
    __syncthreads();

    float v = 0.f;
    for (int i = tid; i < N; i += 256) { float d = xr[i] - mu; v += d * d; }
    red[tid] = v; __syncthreads();
    for (int st = 128; st; st >>= 1) { if (tid < st) red[tid] += red[tid + st]; __syncthreads(); }
    const float rstd = rsqrtf(red[0] * (1.0f / N) + 1e-5f);

    for (int i = tid; i < N; i += 256)
        Y[(size_t)row * N + i] = (xr[i] - mu) * rstd * g[i] + bta[i];
}

__global__ void zero_kernel(float* p, int n)
{
    int i = blockIdx.x * blockDim.x + threadIdx.x;
    if (i < n) p[i] = 0.0f;
}

// ---------------------------------------------------------------------------
extern "C" void kernel_launch(void* const* d_in, const int* in_sizes, int n_in,
                              void* d_out, int out_size, void* d_ws, size_t ws_size,
                              hipStream_t stream)
{
    const float* x            = (const float*)d_in[0];
    const float* w_ih         = (const float*)d_in[1];
    const float* w_hh         = (const float*)d_in[2];
    const float* b_ih         = (const float*)d_in[3];
    const float* b_hh         = (const float*)d_in[4];
    const float* enc_attn_w   = (const float*)d_in[5];
    const float* enc_attn_b   = (const float*)d_in[6];
    const float* enc_out_w    = (const float*)d_in[7];
    const float* enc_out_b    = (const float*)d_in[8];
    const float* enc_ln1_g    = (const float*)d_in[9];
    const float* enc_ln1_b    = (const float*)d_in[10];
    const float* enc_ff1_w    = (const float*)d_in[11];
    const float* enc_ff1_b    = (const float*)d_in[12];
    const float* enc_ff2_w    = (const float*)d_in[13];
    const float* enc_ff2_b    = (const float*)d_in[14];
    const float* enc_ln2_g    = (const float*)d_in[15];
    const float* enc_ln2_b    = (const float*)d_in[16];
    const float* enc_norm_g   = (const float*)d_in[17];
    const float* enc_norm_b   = (const float*)d_in[18];
    const float* dec_sa_w     = (const float*)d_in[19];
    const float* dec_sa_b     = (const float*)d_in[20];
    const float* dec_sa_out_w = (const float*)d_in[21];
    const float* dec_sa_out_b = (const float*)d_in[22];
    const float* dec_ln1_g    = (const float*)d_in[23];
    const float* dec_ln1_b    = (const float*)d_in[24];
    const float* dec_ca_w     = (const float*)d_in[25];
    const float* dec_ca_b     = (const float*)d_in[26];
    const float* dec_ca_out_w = (const float*)d_in[27];
    const float* dec_ca_out_b = (const float*)d_in[28];
    const float* dec_ln2_g    = (const float*)d_in[29];
    const float* dec_ln2_b    = (const float*)d_in[30];
    const float* dec_ff1_w    = (const float*)d_in[31];
    const float* dec_ff1_b    = (const float*)d_in[32];
    const float* dec_ff2_w    = (const float*)d_in[33];
    const float* dec_ff2_b    = (const float*)d_in[34];
    const float* dec_ln3_g    = (const float*)d_in[35];
    const float* dec_ln3_b    = (const float*)d_in[36];
    const float* dec_norm_g   = (const float*)d_in[37];
    const float* dec_norm_b   = (const float*)d_in[38];
    const float* fc_w         = (const float*)d_in[39];
    const float* fc_b         = (const float*)d_in[40];
    (void)in_sizes; (void)n_in; (void)out_size; (void)ws_size;

    const size_t BE = (size_t)B_ * E_;
    float* ws   = (float*)d_ws;
    float* h_a  = ws;            // B*E
    float* h_b  = ws + BE;       // B*E
    float* cbuf = ws + 2 * BE;   // B*E
    float* mem  = ws + 3 * BE;   // B*E
    float* tgt  = ws + 4 * BE;   // B*E
    float* vbuf = ws + 5 * BE;   // B*E   (attention V temp)
    float* fbuf = ws + 6 * BE;   // B*F = 2*B*E (FFN hidden)
    float* part = ws + 8 * BE;   // S*B*F = 4*B*E (split-K partials)
    const int S = 2;             // split-K factor

    const dim3 blk(256);

    zero_kernel<<<dim3((unsigned)(BE / 256)), blk, 0, stream>>>(h_a, (int)BE);
    zero_kernel<<<dim3((unsigned)(BE / 256)), blk, 0, stream>>>(cbuf, (int)BE);

    // ---- LSTM recurrence: one fused kernel per timestep, h ping-pong ----
    float* hin = h_a; float* hout = h_b;
    for (int t = 0; t < T_; ++t) {
        lstm_step_kernel<<<dim3(B_ / 32, E_ / 32), blk, 0, stream>>>(
            x, t, hin, hout, cbuf, w_ih, w_hh, b_ih, b_hh);
        float* tmp = hin; hin = hout; hout = tmp;
    }
    const float* h = hin;  // final hidden state (B, E) — untouched below

    auto gemm = [&](const float* A, const float* W, const float* bias,
                    const float* resid, float* C, int N, int K, int relu) {
        gemm_kernel<<<dim3(B_ / 64, (N + 63) / 64, S), blk, 0, stream>>>(
            A, K, W, part, N, K, K / S);
        combine_kernel<<<dim3((B_ * N + 255) / 256), blk, 0, stream>>>(
            part, bias, resid, C, N, S, relu);
    };
    auto ln = [&](float* X, const float* g, const float* b) {
        ln_kernel<<<dim3(B_), blk, 0, stream>>>(X, g, b, X, E_);
    };

    const size_t EE = (size_t)E_ * E_;

    // ---- Encoder (seq_len==1: attention == V proj + out proj) ----
    const float* src = h;
    for (int l = 0; l < L_; ++l) {
        gemm(src, enc_attn_w + (size_t)l * 3 * EE + 2 * EE,
             enc_attn_b + (size_t)l * 3 * E_ + 2 * E_, nullptr, vbuf, E_, E_, 0);
        gemm(vbuf, enc_out_w + (size_t)l * EE, enc_out_b + (size_t)l * E_,
             src, mem, E_, E_, 0);                                   // + residual
        ln(mem, enc_ln1_g + (size_t)l * E_, enc_ln1_b + (size_t)l * E_);
        gemm(mem, enc_ff1_w + (size_t)l * F_ * E_, enc_ff1_b + (size_t)l * F_,
             nullptr, fbuf, F_, E_, 1);                              // ReLU
        gemm(fbuf, enc_ff2_w + (size_t)l * E_ * F_, enc_ff2_b + (size_t)l * E_,
             mem, mem, E_, F_, 0);                                   // + residual
        ln(mem, enc_ln2_g + (size_t)l * E_, enc_ln2_b + (size_t)l * E_);
        src = mem;
    }
    ln(mem, enc_norm_g, enc_norm_b);

    // ---- Decoder ----
    const float* tsrc = h;
    for (int l = 0; l < L_; ++l) {
        gemm(tsrc, dec_sa_w + (size_t)l * 3 * EE + 2 * EE,
             dec_sa_b + (size_t)l * 3 * E_ + 2 * E_, nullptr, vbuf, E_, E_, 0);
        gemm(vbuf, dec_sa_out_w + (size_t)l * EE, dec_sa_out_b + (size_t)l * E_,
             tsrc, tgt, E_, E_, 0);
        ln(tgt, dec_ln1_g + (size_t)l * E_, dec_ln1_b + (size_t)l * E_);
        gemm(mem, dec_ca_w + (size_t)l * 3 * EE + 2 * EE,
             dec_ca_b + (size_t)l * 3 * E_ + 2 * E_, nullptr, vbuf, E_, E_, 0);
        gemm(vbuf, dec_ca_out_w + (size_t)l * EE, dec_ca_out_b + (size_t)l * E_,
             tgt, tgt, E_, E_, 0);
        ln(tgt, dec_ln2_g + (size_t)l * E_, dec_ln2_b + (size_t)l * E_);
        gemm(tgt, dec_ff1_w + (size_t)l * F_ * E_, dec_ff1_b + (size_t)l * F_,
             nullptr, fbuf, F_, E_, 1);
        gemm(fbuf, dec_ff2_w + (size_t)l * E_ * F_, dec_ff2_b + (size_t)l * E_,
             tgt, tgt, E_, F_, 0);
        ln(tgt, dec_ln3_g + (size_t)l * E_, dec_ln3_b + (size_t)l * E_);
        tsrc = tgt;
    }
    ln(tgt, dec_norm_g, dec_norm_b);

    // ---- Head: (B,E) @ (O,E)^T + b -> d_out (f32, 256x22) ----
    gemm(tgt, fc_w, fc_b, nullptr, (float*)d_out, O_, E_, 0);
}
